// NonLocalBlock_14577119002790
// MI455X (gfx1250) — compile-verified, matrix operand order checked by reading, not run on hardware
//
#include <hip/hip_runtime.h>

// ---------------------------------------------------------------------------
// Types
// ---------------------------------------------------------------------------
typedef __bf16  bf16x16 __attribute__((ext_vector_type(16)));
typedef __bf16  bf16x8  __attribute__((ext_vector_type(8)));
typedef __bf16  bf16x4  __attribute__((ext_vector_type(4)));
typedef float   f32x8   __attribute__((ext_vector_type(8)));
typedef float   f32x4   __attribute__((ext_vector_type(4)));

#define C_DIM 512
#define N_DIM 4096   // 64*64
#define B_DIM 4

// ---------------------------------------------------------------------------
// WMMA helpers (CDNA5 v_wmma_f32_16x16x32_bf16, wave32)
// ---------------------------------------------------------------------------
__device__ __forceinline__ f32x8 wmma_bf16(bf16x16 a, bf16x16 b, f32x8 c) {
  return __builtin_amdgcn_wmma_f32_16x16x32_bf16(false, a, false, b, (short)0, c,
                                                 false, false);
}

// Fragment load for A ([M][K] memory) or B ([N][K] memory) when the K dim is
// contiguous. Per the ISA 16-bit layout, lane L needs row (L&15) with two
// contiguous 8-element K runs at K = kh and K = 16+kh, kh = 8*(L>=16):
// exactly two 16-byte loads per lane.
__device__ __forceinline__ bf16x16 frag_ld(const __bf16* p, int ld) {
  const int lane = threadIdx.x & 31;
  const __bf16* row = p + (size_t)(lane & 15) * (size_t)ld + ((lane >> 4) << 3);
  bf16x8 lo = *(const bf16x8*)row;
  bf16x8 hi = *(const bf16x8*)(row + 16);
  bf16x16 f;
#pragma unroll
  for (int e = 0; e < 8; ++e) {
    f[e] = lo[e];
    f[e + 8] = hi[e];
  }
  return f;
}

// ---------------------------------------------------------------------------
// GroupNorm stats: one block per (group, batch)
// ---------------------------------------------------------------------------
__global__ void __launch_bounds__(256) gn_stats(const float* __restrict__ x,
                                                float* __restrict__ mean,
                                                float* __restrict__ rstd) {
  const int g = blockIdx.x;   // 32 groups
  const int z = blockIdx.y;   // batch
  const float* base = x + (((size_t)z * C_DIM) + (size_t)g * 16) * N_DIM;
  float s = 0.f, ss = 0.f;
  for (int i = threadIdx.x; i < 16 * N_DIM; i += 256) {
    float v = base[i];
    s += v;
    ss += v * v;
  }
#pragma unroll
  for (int mask = 16; mask >= 1; mask >>= 1) {
    s  += __shfl_xor(s,  mask, 32);
    ss += __shfl_xor(ss, mask, 32);
  }
  __shared__ float rs[8], rss[8];
  if ((threadIdx.x & 31) == 0) {
    rs[threadIdx.x >> 5]  = s;
    rss[threadIdx.x >> 5] = ss;
  }
  __syncthreads();
  if (threadIdx.x == 0) {
    float S = 0.f, SS = 0.f;
#pragma unroll
    for (int w = 0; w < 8; ++w) { S += rs[w]; SS += rss[w]; }
    const float inv_n = 1.0f / (16.0f * (float)N_DIM);
    float mu  = S * inv_n;
    float var = SS * inv_n - mu * mu;
    mean[z * 32 + g] = mu;
    rstd[z * 32 + g] = rsqrtf(var + 1e-6f);
  }
}

// GroupNorm apply + bf16 convert, writing y TRANSPOSED: yT[b][n][c]
// so GEMM B-fragments (K = c) are contiguous 16B loads.
__global__ void __launch_bounds__(256) gn_apply(const float* __restrict__ x,
                                                const float* __restrict__ mean,
                                                const float* __restrict__ rstd,
                                                const float* __restrict__ gs,
                                                const float* __restrict__ gb,
                                                __bf16* __restrict__ yT) {
  size_t i4 = (size_t)blockIdx.x * 256 + threadIdx.x;
  size_t base = i4 * 4;
  int j = (int)(base & (N_DIM - 1));
  int c = (int)((base >> 12) & (C_DIM - 1));
  int z = (int)(base >> 21);
  int gi = z * 32 + (c >> 4);
  float mu = mean[gi], rv = rstd[gi];
  float sc = gs[c] * rv;
  float bi = gb[c] - mu * sc;
  f32x4 v = *(const f32x4*)&x[base];
#pragma unroll
  for (int e = 0; e < 4; ++e)
    yT[((size_t)z * N_DIM + j + e) * C_DIM + c] = (__bf16)(v[e] * sc + bi);
}

// ---------------------------------------------------------------------------
// Convert the 4 weight matrices (512x512 f32) to bf16
// ---------------------------------------------------------------------------
__global__ void __launch_bounds__(256) wconv(const float* __restrict__ a,
                                             const float* __restrict__ b,
                                             const float* __restrict__ c,
                                             const float* __restrict__ d,
                                             __bf16* __restrict__ oa,
                                             __bf16* __restrict__ ob,
                                             __bf16* __restrict__ oc,
                                             __bf16* __restrict__ od) {
  size_t i = ((size_t)blockIdx.x * 256 + threadIdx.x) * 4;
  f32x4 va = *(const f32x4*)&a[i];
  f32x4 vb = *(const f32x4*)&b[i];
  f32x4 vc = *(const f32x4*)&c[i];
  f32x4 vd = *(const f32x4*)&d[i];
  bf16x4 ra, rb, rc, rd;
#pragma unroll
  for (int e = 0; e < 4; ++e) {
    ra[e] = (__bf16)va[e]; rb[e] = (__bf16)vb[e];
    rc[e] = (__bf16)vc[e]; rd[e] = (__bf16)vd[e];
  }
  *(bf16x4*)&oa[i] = ra; *(bf16x4*)&ob[i] = rb;
  *(bf16x4*)&oc[i] = rc; *(bf16x4*)&od[i] = rd;
}

// ---------------------------------------------------------------------------
// WMMA GEMM:  out[o, j] = sum_i W[o,i] * ActT[j,i] + bias[o]
//  W:    [512 o][512 i] bf16  (A fragments: K=i contiguous)
//  ActT: [b][4096 j][512 i]   (B fragments: K=i contiguous)
//  -> no LDS, no barriers; operands stream through L2 (192 MB).
//  8 waves/block, block tile 128(M) x 64(N), wave tile 32x32, K-step 32.
//  MODE 0: bf16 out [c][n]          (V)
//  MODE 1: bf16 out [n][c]          (Q^T, K^T)
//  MODE 2: f32 out [c][n] + residual (final projection -> d_out)
// ---------------------------------------------------------------------------
template <int MODE>
__global__ void __launch_bounds__(256) gemm_wmma(const __bf16* __restrict__ W,
                                                 const __bf16* __restrict__ ActT,
                                                 const float* __restrict__ bias,
                                                 const float* __restrict__ resid,
                                                 __bf16* __restrict__ outb,
                                                 float* __restrict__ outf) {
  const int tid = threadIdx.x;
  const int wid = tid >> 5;
  const int jb = blockIdx.x * 64;    // spatial tile
  const int ob = blockIdx.y * 128;   // out-channel tile
  const int z  = blockIdx.z;         // batch
  const int wm = (wid & 3) * 32;     // wave M offset
  const int wn = (wid >> 2) * 32;    // wave N offset
  f32x8 acc[2][2] = {};
  const __bf16* Arow0 = W + (size_t)(ob + wm) * C_DIM;
  const __bf16* Arow1 = W + (size_t)(ob + wm + 16) * C_DIM;
  const __bf16* Brow0 = ActT + ((size_t)z * N_DIM + jb + wn) * C_DIM;
  const __bf16* Brow1 = ActT + ((size_t)z * N_DIM + jb + wn + 16) * C_DIM;

#pragma unroll 4
  for (int k0 = 0; k0 < C_DIM; k0 += 32) {
    if (k0 + 32 < C_DIM) {  // global_prefetch_b8 for next K slab
      __builtin_prefetch(Arow0 + k0 + 32, 0, 3);
      __builtin_prefetch(Brow0 + k0 + 32, 0, 3);
    }
    bf16x16 a0 = frag_ld(Arow0 + k0, C_DIM);
    bf16x16 a1 = frag_ld(Arow1 + k0, C_DIM);
    bf16x16 b0 = frag_ld(Brow0 + k0, C_DIM);
    bf16x16 b1 = frag_ld(Brow1 + k0, C_DIM);
    acc[0][0] = wmma_bf16(a0, b0, acc[0][0]);
    acc[0][1] = wmma_bf16(a0, b1, acc[0][1]);
    acc[1][0] = wmma_bf16(a1, b0, acc[1][0]);
    acc[1][1] = wmma_bf16(a1, b1, acc[1][1]);
  }

  // epilogue: D layout -> lane nn = N col, M = r + 8*(lane>=16)
  const int lane = tid & 31, nn = lane & 15, mh = (lane >> 4) << 3;
#pragma unroll
  for (int i = 0; i < 2; ++i)
#pragma unroll
    for (int jj = 0; jj < 2; ++jj) {
      int o0 = ob + wm + i * 16;
      int j0 = jb + wn + jj * 16 + nn;
#pragma unroll
      for (int r = 0; r < 8; ++r) {
        int oo = o0 + mh + r;
        float val = acc[i][jj][r] + bias[oo];
        if (MODE == 0) {
          outb[((size_t)z * C_DIM + oo) * N_DIM + j0] = (__bf16)val;
        } else if (MODE == 1) {
          outb[((size_t)z * N_DIM + j0) * C_DIM + oo] = (__bf16)val;
        } else {
          size_t idx = ((size_t)z * C_DIM + oo) * N_DIM + j0;
          outf[idx] = resid[idx] + val;
        }
      }
    }
}

// ---------------------------------------------------------------------------
// Fused flash attention, n=4096 keys, head dim 512.
//  qT, kT: [b][n][c]  (fragments contiguous along c)
//  V:      [b][c][n]  (P x V^T fragments contiguous along n)
//  oT out: [b][n][c]  (feeds projection GEMM directly)
//  Block = 8 waves, 16 query rows. Score c-reduction split across waves,
//  merged in LDS with ds_add_f32. Q fragments hoisted out of the j-loop.
// ---------------------------------------------------------------------------
__global__ void __launch_bounds__(256) flash_attn(const __bf16* __restrict__ qT,
                                                  const __bf16* __restrict__ kT,
                                                  const __bf16* __restrict__ V,
                                                  __bf16* __restrict__ oT) {
  __shared__ __attribute__((aligned(16))) float  sS[16 * 32];  // score merge
  __shared__ __attribute__((aligned(16))) __bf16 sP[16 * 32];  // P repack

  const int tid = threadIdx.x, wid = tid >> 5, lane = tid & 31;
  const int nn = lane & 15, mh = (lane >> 4) << 3;
  const int ib = blockIdx.x * 16;  // query tile
  const int z  = blockIdx.y;       // batch
  const float scl = 0.044194173824159216f;  // 512^-0.5

  const __bf16* Qz = qT + ((size_t)z * N_DIM + ib) * C_DIM;
  const __bf16* Kz = kT + (size_t)z * N_DIM * C_DIM;
  const __bf16* Vz = V + (size_t)z * C_DIM * N_DIM;

  // Q fragments for this wave's two c-chunks are j-invariant: hoist.
  const int cc0 = wid * 64;
  const int cc1 = wid * 64 + 32;
  const bf16x16 aq0 = frag_ld(Qz + cc0, C_DIM);
  const bf16x16 aq1 = frag_ld(Qz + cc1, C_DIM);

  float m_i[8], l_i[8];
  f32x8 oacc[4] = {};
#pragma unroll
  for (int r = 0; r < 8; ++r) { m_i[r] = -1e30f; l_i[r] = 0.f; }

  for (int j0 = 0; j0 < N_DIM; j0 += 32) {
    sS[tid] = 0.f;
    sS[tid + 256] = 0.f;

    // prefetch next j-step's K rows and V rows (global_prefetch_b8)
    if (j0 + 32 < N_DIM) {
      __builtin_prefetch(Kz + (size_t)(j0 + 32 + (lane & 15)) * C_DIM + cc0, 0, 3);
      __builtin_prefetch(Vz + (size_t)(cc0 + (lane & 15)) * N_DIM + j0 + 32, 0, 3);
    }
    __syncthreads();  // B1

    // partial scores: issue all four K fragments, then the WMMAs
    bf16x16 bk00 = frag_ld(Kz + (size_t)j0 * C_DIM + cc0, C_DIM);
    bf16x16 bk10 = frag_ld(Kz + (size_t)(j0 + 16) * C_DIM + cc0, C_DIM);
    bf16x16 bk01 = frag_ld(Kz + (size_t)j0 * C_DIM + cc1, C_DIM);
    bf16x16 bk11 = frag_ld(Kz + (size_t)(j0 + 16) * C_DIM + cc1, C_DIM);
    f32x8 s0 = {}, s1 = {};
    s0 = wmma_bf16(aq0, bk00, s0);
    s1 = wmma_bf16(aq0, bk10, s1);
    s0 = wmma_bf16(aq1, bk01, s0);
    s1 = wmma_bf16(aq1, bk11, s1);

#pragma unroll
    for (int r = 0; r < 8; ++r) {  // merge partials across waves: ds_add_f32
      int m = mh + r;
      atomicAdd(&sS[m * 32 + nn], s0[r]);
      atomicAdd(&sS[m * 32 + 16 + nn], s1[r]);
    }
    __syncthreads();  // B2

    // online softmax update (replicated per wave; identical results)
    float p0[8], p1[8];
#pragma unroll
    for (int r = 0; r < 8; ++r) {
      int m = mh + r;
      float v0 = sS[m * 32 + nn] * scl;
      float v1 = sS[m * 32 + 16 + nn] * scl;
      float rmax = fmaxf(v0, v1);
#pragma unroll
      for (int mask = 1; mask <= 8; mask <<= 1)
        rmax = fmaxf(rmax, __shfl_xor(rmax, mask, 32));
      float mnew = fmaxf(m_i[r], rmax);
      float corr = __expf(m_i[r] - mnew);
      float e0 = __expf(v0 - mnew);
      float e1 = __expf(v1 - mnew);
      float rsum = e0 + e1;
#pragma unroll
      for (int mask = 1; mask <= 8; mask <<= 1)
        rsum += __shfl_xor(rsum, mask, 32);
      l_i[r] = l_i[r] * corr + rsum;
      m_i[r] = mnew;
      p0[r] = e0;
      p1[r] = e1;
#pragma unroll
      for (int t = 0; t < 4; ++t) oacc[t][r] *= corr;
    }
    if (wid == 0) {  // publish P (bf16) in [M=16][K=32] layout
#pragma unroll
      for (int r = 0; r < 8; ++r) {
        int m = mh + r;
        sP[m * 32 + nn]      = (__bf16)p0[r];
        sP[m * 32 + 16 + nn] = (__bf16)p1[r];
      }
    }
    __syncthreads();  // B3

    // O += P x V^T for this wave's 64 channels
    bf16x16 pA  = frag_ld(sP, 32);
    bf16x16 vB0 = frag_ld(Vz + (size_t)(cc0) * N_DIM + j0, N_DIM);
    bf16x16 vB1 = frag_ld(Vz + (size_t)(cc0 + 16) * N_DIM + j0, N_DIM);
    bf16x16 vB2 = frag_ld(Vz + (size_t)(cc0 + 32) * N_DIM + j0, N_DIM);
    bf16x16 vB3 = frag_ld(Vz + (size_t)(cc0 + 48) * N_DIM + j0, N_DIM);
    oacc[0] = wmma_bf16(pA, vB0, oacc[0]);
    oacc[1] = wmma_bf16(pA, vB1, oacc[1]);
    oacc[2] = wmma_bf16(pA, vB2, oacc[2]);
    oacc[3] = wmma_bf16(pA, vB3, oacc[3]);
  }

  // epilogue: oT[b][i][c] = acc / l  (transposed out for projection GEMM)
#pragma unroll
  for (int t = 0; t < 4; ++t) {
    int cb = cc0 + t * 16;
#pragma unroll
    for (int r = 0; r < 8; ++r) {
      float val = oacc[t][r] / l_i[r];
      oT[((size_t)z * N_DIM + ib + mh + r) * C_DIM + cb + nn] = (__bf16)val;
    }
  }
}

// ---------------------------------------------------------------------------
// Host launcher
// ---------------------------------------------------------------------------
extern "C" void kernel_launch(void* const* d_in, const int* in_sizes, int n_in,
                              void* d_out, int out_size, void* d_ws,
                              size_t ws_size, hipStream_t stream) {
  const float* x   = (const float*)d_in[0];
  const float* gsc = (const float*)d_in[1];
  const float* gbi = (const float*)d_in[2];
  const float* qw  = (const float*)d_in[3];
  const float* qb  = (const float*)d_in[4];
  const float* kw  = (const float*)d_in[5];
  const float* kb  = (const float*)d_in[6];
  const float* vw  = (const float*)d_in[7];
  const float* vb  = (const float*)d_in[8];
  const float* pw  = (const float*)d_in[9];
  const float* pb  = (const float*)d_in[10];
  float* out = (float*)d_out;

  // workspace carve-out (256B aligned slabs)
  char* wp_ = (char*)d_ws;
  auto carve = [&](size_t bytes) -> void* {
    void* p = (void*)wp_;
    wp_ += (bytes + 255) & ~(size_t)255;
    return p;
  };
  const size_t act = (size_t)B_DIM * C_DIM * N_DIM;  // 8388608 elems
  const size_t wsz = (size_t)C_DIM * C_DIM;          // 262144 elems
  float*  mean = (float*)carve(B_DIM * 32 * sizeof(float));
  float*  rstd = (float*)carve(B_DIM * 32 * sizeof(float));
  __bf16* wqb  = (__bf16*)carve(wsz * 2);
  __bf16* wkb  = (__bf16*)carve(wsz * 2);
  __bf16* wvb  = (__bf16*)carve(wsz * 2);
  __bf16* wpb  = (__bf16*)carve(wsz * 2);
  __bf16* yT   = (__bf16*)carve(act * 2);
  __bf16* qT   = (__bf16*)carve(act * 2);
  __bf16* kT   = (__bf16*)carve(act * 2);
  __bf16* vv   = (__bf16*)carve(act * 2);
  __bf16* oT   = (__bf16*)carve(act * 2);

  gn_stats<<<dim3(32, B_DIM), 256, 0, stream>>>(x, mean, rstd);
  gn_apply<<<(int)(act / 4 / 256), 256, 0, stream>>>(x, mean, rstd, gsc, gbi,
                                                     yT);
  wconv<<<(int)(wsz / 4 / 256), 256, 0, stream>>>(qw, kw, vw, pw, wqb, wkb, wvb,
                                                  wpb);

  dim3 gg(N_DIM / 64, C_DIM / 128, B_DIM);
  gemm_wmma<1><<<gg, 256, 0, stream>>>(wqb, yT, qb, nullptr, qT, nullptr);
  gemm_wmma<1><<<gg, 256, 0, stream>>>(wkb, yT, kb, nullptr, kT, nullptr);
  gemm_wmma<0><<<gg, 256, 0, stream>>>(wvb, yT, vb, nullptr, vv, nullptr);

  flash_attn<<<dim3(N_DIM / 16, B_DIM), 256, 0, stream>>>(qT, kT, vv, oT);

  gemm_wmma<2><<<gg, 256, 0, stream>>>(wpb, oT, pb, x, nullptr, out);
}
